// KolmogorovArnoldNetwork_87660282511837
// MI455X (gfx1250) — compile-verified, matrix operand order checked by reading, not run on hardware
//
#include <hip/hip_runtime.h>
#include <cstdint>

// Problem constants (from reference): B=2048, P=32, Q=65, O=16, H=32
static constexpr int Bc = 2048;
static constexpr int Pc = 32;
static constexpr int Qc = 65;
static constexpr int Oc = 16;
static constexpr int Hc = 32;

typedef __attribute__((ext_vector_type(16))) _Float16 v16h;
typedef __attribute__((ext_vector_type(8)))  float    v8f;

__device__ __forceinline__ uint32_t pack_half2(_Float16 a, _Float16 b) {
    union { _Float16 h[2]; uint32_t u; } u_;
    u_.h[0] = a; u_.h[1] = b;
    return u_.u;
}

__device__ __forceinline__ float fast_tanh(float x) {
#if __has_builtin(__builtin_amdgcn_tanhf)
    return __builtin_amdgcn_tanhf(x);
#elif __has_builtin(__builtin_amdgcn_tanh_f32)
    return __builtin_amdgcn_tanh_f32(x);
#else
    float xc = fminf(fmaxf(x, -15.0f), 15.0f);
    float e  = __builtin_amdgcn_exp2f(xc * 2.8853900817779268f);
    return (e - 1.0f) * __builtin_amdgcn_rcpf(e + 1.0f);
#endif
}

__device__ __forceinline__ _Float16 tanh_h(_Float16 x) {
#if __has_builtin(__builtin_amdgcn_tanh_f16)
    return __builtin_amdgcn_tanh_f16(x);
#elif __has_builtin(__builtin_amdgcn_tanhh)
    return __builtin_amdgcn_tanhh(x);
#else
    return (_Float16)fast_tanh((float)x);
#endif
}

// ---------------------------------------------------------------------------
// Pre-swizzle [H,H] f32 weight blocks into the v_wmma_f32_16x16x32_f16
// B-matrix fragment layout (f16), one 16x16-output N-tile per "tile".
// Tile t: pq = t>>1, ntile = t&1.  Lane l: n=(l&15)+16*ntile, kg=l>>4.
// Element e of the v16h: K = 16*(e>>3) + 8*kg + (e&7)   (ISA 7.12.2)
// ---------------------------------------------------------------------------
__global__ __launch_bounds__(32) void swizzle_w2_f16(
    const float* __restrict__ src, uint32_t* __restrict__ dst)
{
    const int t    = blockIdx.x;
    const int lane = threadIdx.x;
    const int pq   = t >> 1;
    const int nt   = t & 1;
    const int n    = (lane & 15) + nt * 16;
    const int kg   = lane >> 4;
    const float* w = src + (size_t)pq * (Hc * Hc);

    uint32_t out[8];
#pragma unroll
    for (int j = 0; j < 8; ++j) {
        const int e0 = 2 * j, e1 = e0 + 1;
        const int K0 = ((e0 >> 3) << 4) + kg * 8 + (e0 & 7);
        const int K1 = ((e1 >> 3) << 4) + kg * 8 + (e1 & 7);
        out[j] = pack_half2((_Float16)w[K0 * Hc + n], (_Float16)w[K1 * Hc + n]);
    }
    uint32_t* d = dst + (size_t)t * 256 + lane * 8;
#pragma unroll
    for (int j = 0; j < 8; ++j) d[j] = out[j];
}

// ---------------------------------------------------------------------------
// Pre-swizzle layer-1 weight+bias vectors [npq,H] into packed-f16 A-fragment
// K-order.  Output layout per pq (32 dwords):
//   [kg=0: w1(8 dwords) b1(8 dwords)][kg=1: w1(8) b1(8)]
// ---------------------------------------------------------------------------
__global__ __launch_bounds__(256) void swizzle_w1b1_f16(
    const float* __restrict__ w1, const float* __restrict__ b1,
    uint32_t* __restrict__ dst, int npq)
{
    const int idx = blockIdx.x * 256 + threadIdx.x;
    if (idx >= npq * 32) return;
    const int pq   = idx >> 5;
    const int r    = idx & 31;
    const int kg   = r >> 4;
    const int half = (r >> 3) & 1;
    const int j    = r & 7;
    const float* src = (half == 0 ? w1 : b1) + (size_t)pq * Hc;
    const int e0 = 2 * j, e1 = e0 + 1;
    const int K0 = ((e0 >> 3) << 4) + kg * 8 + (e0 & 7);
    const int K1 = ((e1 >> 3) << 4) + kg * 8 + (e1 & 7);
    dst[idx] = pack_half2((_Float16)src[K0], (_Float16)src[K1]);
}

// ---------------------------------------------------------------------------
// Inner psi pass: s[q,b] = sum_p ( w3 . tanh(W2 tanh(x*w1+b1) + b2) + b3 )
// Grid: (Q, B/64).  Block: 128 threads = 4 waves, 16 batch rows per wave.
// Software-pipelined: next iteration's A fragment (pk_fma_f16 + tanh_f16,
// all TRANS ops) is computed between the WMMAs and their consumers to fill
// the WMMA->VALU hazard window.  ONE cross-lane reduction at the end.
// ---------------------------------------------------------------------------
__global__ __launch_bounds__(128) void psi_kernel(
    const float* __restrict__ x,        // [B,P]
    const uint32_t* __restrict__ w1b1s, // swizzled f16 [P*Q*32]
    const uint32_t* __restrict__ w2s,   // swizzled f16 [P*Q*2*256]
    const float* __restrict__ b2,       // [P,Q,H]
    const float* __restrict__ w3,       // [P,Q,H]
    const float* __restrict__ b3,       // [P,Q]
    float* __restrict__ s_out)          // [Q,B]
{
    const int q    = blockIdx.x;
    const int wave = threadIdx.x >> 5;
    const int lane = threadIdx.x & 31;
    const int m    = lane & 15;        // A row / D column index
    const int kg   = lane >> 4;        // K-group / D row-group
    const int rowbase = (blockIdx.y * 4 + wave) * 16;
    const int arow = rowbase + m;

    float acc[8];
#pragma unroll
    for (int i = 0; i < 8; ++i) acc[i] = 0.0f;
    float bsum = 0.0f;

    // ---- Prologue: stage p = 0 ----
    v16h a_cur, bf0, bf1;
    float b2a, b2b, w3a, w3b, b3v;
    {
        const int pq = q;  // p = 0
        const float xv = x[(size_t)arow * Pc + 0];
        const uint32_t* wb = w1b1s + (size_t)pq * 32 + kg * 16;
        v16h wf  = *(const v16h*)(wb);
        v16h bfv = *(const v16h*)(wb + 8);
        v16h h1 = wf * (_Float16)xv + bfv;
#pragma unroll
        for (int e = 0; e < 16; ++e) a_cur[e] = tanh_h(h1[e]);
        const uint32_t* bp = w2s + (size_t)(pq * 2) * 256 + lane * 8;
        bf0 = *(const v16h*)(bp);
        bf1 = *(const v16h*)(bp + 256);
        b2a = b2[(size_t)pq * Hc + m];
        b2b = b2[(size_t)pq * Hc + m + 16];
        w3a = w3[(size_t)pq * Hc + m];
        w3b = w3[(size_t)pq * Hc + m + 16];
        b3v = b3[pq];
    }

#pragma unroll 2
    for (int p = 0; p < Pc; ++p) {
        v8f c0 = {}; v8f c1 = {};
        c0 = __builtin_amdgcn_wmma_f32_16x16x32_f16(false, a_cur, false, bf0,
                                                    (short)0, c0, false, false);
        c1 = __builtin_amdgcn_wmma_f32_16x16x32_f16(false, a_cur, false, bf1,
                                                    (short)0, c1, false, false);
        const float eb2a = b2a, eb2b = b2b, ew3a = w3a, ew3b = w3b;
        bsum += b3v;

        // Stage p+1: loads + A-fragment TRANS work fill the hazard window.
        v16h a_nxt = a_cur;
        if (p + 1 < Pc) {
            const int pqn = (p + 1) * Qc + q;
            const float xv = x[(size_t)arow * Pc + p + 1];
            const uint32_t* wbn = w1b1s + (size_t)pqn * 32 + kg * 16;
            v16h wf  = *(const v16h*)(wbn);
            v16h bfv = *(const v16h*)(wbn + 8);
            const uint32_t* bpn = w2s + (size_t)(pqn * 2) * 256 + lane * 8;
            bf0 = *(const v16h*)(bpn);
            bf1 = *(const v16h*)(bpn + 256);
            if (p + 2 < Pc)
                __builtin_prefetch(w2s + (size_t)(((p + 2) * Qc + q) * 2) * 256 + lane * 8, 0, 0);
            b2a = b2[(size_t)pqn * Hc + m];
            b2b = b2[(size_t)pqn * Hc + m + 16];
            w3a = w3[(size_t)pqn * Hc + m];
            w3b = w3[(size_t)pqn * Hc + m + 16];
            b3v = b3[pqn];
            v16h h1 = wf * (_Float16)xv + bfv;
#pragma unroll
            for (int e = 0; e < 16; ++e) a_nxt[e] = tanh_h(h1[e]);
        }

        // Consume WMMA results: per-lane partial accumulators (no reduction).
#pragma unroll
        for (int i = 0; i < 8; ++i)
            acc[i] += fast_tanh(c0[i] + eb2a) * ew3a + fast_tanh(c1[i] + eb2b) * ew3b;
        a_cur = a_nxt;
    }

    // Single cross-lane reduction over the 16 columns of each half.
#pragma unroll
    for (int mask = 1; mask <= 8; mask <<= 1)
#pragma unroll
        for (int i = 0; i < 8; ++i)
            acc[i] += __shfl_xor(acc[i], mask, 32);

    if (m == 0) {
        const int rb = rowbase + kg * 8;
#pragma unroll
        for (int i = 0; i < 8; ++i)
            s_out[(size_t)q * Bc + rb + i] = acc[i] + bsum;  // coalesced [Q,B]
    }
}

// ---------------------------------------------------------------------------
// Outer phi pass: out[b,o] += sum_q ( w3 . tanh(W2 tanh(s*w1+b1) + b2) + b3 )
// Grid: (B/64, 8 q-chunks).  o-outer / q-inner (one reduction per o), with
// the q-loop software-pipelined like psi.  Accumulates via f32 atomics.
// ---------------------------------------------------------------------------
__global__ __launch_bounds__(128) void phi_kernel(
    const float* __restrict__ s_in,     // [Q,B]
    const uint32_t* __restrict__ w1b1s, // swizzled f16 [Q*O*32]
    const uint32_t* __restrict__ w2s,   // swizzled f16 [Q*O*2*256]
    const float* __restrict__ b2,       // [Q,O,H]
    const float* __restrict__ w3,       // [Q,O,H]
    const float* __restrict__ b3,       // [Q,O]
    float* __restrict__ out)            // [B,O]
{
    const int wave = threadIdx.x >> 5;
    const int lane = threadIdx.x & 31;
    const int m    = lane & 15;
    const int kg   = lane >> 4;
    const int rowbase = (blockIdx.x * 4 + wave) * 16;
    const int arow = rowbase + m;
    const int q0 = blockIdx.y * 9;
    const int q1 = (q0 + 9 < Qc) ? (q0 + 9) : Qc;
    const int rb = rowbase + kg * 8;

    for (int o = 0; o < Oc; ++o) {
        float acc[8];
#pragma unroll
        for (int i = 0; i < 8; ++i) acc[i] = 0.0f;
        float bsum = 0.0f;

        // ---- Prologue: stage q = q0 ----
        v16h a_cur, bf0, bf1;
        float b2a, b2b, w3a, w3b, b3v;
        {
            const int qo = q0 * Oc + o;
            const float sv = s_in[(size_t)q0 * Bc + arow];
            const uint32_t* wb = w1b1s + (size_t)qo * 32 + kg * 16;
            v16h wf  = *(const v16h*)(wb);
            v16h bfv = *(const v16h*)(wb + 8);
            v16h h1 = wf * (_Float16)sv + bfv;
#pragma unroll
            for (int e = 0; e < 16; ++e) a_cur[e] = tanh_h(h1[e]);
            const uint32_t* bp = w2s + (size_t)(qo * 2) * 256 + lane * 8;
            bf0 = *(const v16h*)(bp);
            bf1 = *(const v16h*)(bp + 256);
            b2a = b2[(size_t)qo * Hc + m];
            b2b = b2[(size_t)qo * Hc + m + 16];
            w3a = w3[(size_t)qo * Hc + m];
            w3b = w3[(size_t)qo * Hc + m + 16];
            b3v = b3[qo];
        }

#pragma unroll 2
        for (int q = q0; q < q1; ++q) {
            v8f c0 = {}; v8f c1 = {};
            c0 = __builtin_amdgcn_wmma_f32_16x16x32_f16(false, a_cur, false, bf0,
                                                        (short)0, c0, false, false);
            c1 = __builtin_amdgcn_wmma_f32_16x16x32_f16(false, a_cur, false, bf1,
                                                        (short)0, c1, false, false);
            const float eb2a = b2a, eb2b = b2b, ew3a = w3a, ew3b = w3b;
            bsum += b3v;

            v16h a_nxt = a_cur;
            if (q + 1 < q1) {
                const int qon = (q + 1) * Oc + o;
                const float sv = s_in[(size_t)(q + 1) * Bc + arow];
                const uint32_t* wbn = w1b1s + (size_t)qon * 32 + kg * 16;
                v16h wf  = *(const v16h*)(wbn);
                v16h bfv = *(const v16h*)(wbn + 8);
                const uint32_t* bpn = w2s + (size_t)(qon * 2) * 256 + lane * 8;
                bf0 = *(const v16h*)(bpn);
                bf1 = *(const v16h*)(bpn + 256);
                b2a = b2[(size_t)qon * Hc + m];
                b2b = b2[(size_t)qon * Hc + m + 16];
                w3a = w3[(size_t)qon * Hc + m];
                w3b = w3[(size_t)qon * Hc + m + 16];
                b3v = b3[qon];
                v16h h1 = wf * (_Float16)sv + bfv;
#pragma unroll
                for (int e = 0; e < 16; ++e) a_nxt[e] = tanh_h(h1[e]);
            }

#pragma unroll
            for (int i = 0; i < 8; ++i)
                acc[i] += fast_tanh(c0[i] + eb2a) * ew3a + fast_tanh(c1[i] + eb2b) * ew3b;
            a_cur = a_nxt;
        }

        // One reduction per output column o.
#pragma unroll
        for (int mask = 1; mask <= 8; mask <<= 1)
#pragma unroll
            for (int i = 0; i < 8; ++i)
                acc[i] += __shfl_xor(acc[i], mask, 32);

        if (m == 0) {
#pragma unroll
            for (int i = 0; i < 8; ++i)
                atomicAdd(&out[(size_t)(rb + i) * Oc + o], acc[i] + bsum);
        }
    }
}

// ---------------------------------------------------------------------------
extern "C" void kernel_launch(void* const* d_in, const int* in_sizes, int n_in,
                              void* d_out, int out_size, void* d_ws, size_t ws_size,
                              hipStream_t stream)
{
    (void)in_sizes; (void)n_in; (void)out_size; (void)ws_size;
    const float* x      = (const float*)d_in[0];
    const float* psi_w1 = (const float*)d_in[1];
    const float* psi_b1 = (const float*)d_in[2];
    const float* psi_w2 = (const float*)d_in[3];
    const float* psi_b2 = (const float*)d_in[4];
    const float* psi_w3 = (const float*)d_in[5];
    const float* psi_b3 = (const float*)d_in[6];
    const float* phi_w1 = (const float*)d_in[7];
    const float* phi_b1 = (const float*)d_in[8];
    const float* phi_w2 = (const float*)d_in[9];
    const float* phi_b2 = (const float*)d_in[10];
    const float* phi_w3 = (const float*)d_in[11];
    const float* phi_b3 = (const float*)d_in[12];

    // Workspace layout
    uint32_t* psi_w2s   = (uint32_t*)d_ws;                              // P*Q*2*256 dw
    uint32_t* phi_w2s   = psi_w2s + (size_t)Pc * Qc * 2 * 256;          // Q*O*2*256 dw
    uint32_t* psi_w1b1s = phi_w2s + (size_t)Qc * Oc * 2 * 256;          // P*Q*32 dw
    uint32_t* phi_w1b1s = psi_w1b1s + (size_t)Pc * Qc * 32;             // Q*O*32 dw
    float*    s_buf     = (float*)(phi_w1b1s + (size_t)Qc * Oc * 32);   // Q*B f32

    swizzle_w2_f16<<<Pc * Qc * 2, 32, 0, stream>>>(psi_w2, psi_w2s);
    swizzle_w2_f16<<<Qc * Oc * 2, 32, 0, stream>>>(phi_w2, phi_w2s);
    swizzle_w1b1_f16<<<(Pc * Qc * 32 + 255) / 256, 256, 0, stream>>>(
        psi_w1, psi_b1, psi_w1b1s, Pc * Qc);
    swizzle_w1b1_f16<<<(Qc * Oc * 32 + 255) / 256, 256, 0, stream>>>(
        phi_w1, phi_b1, phi_w1b1s, Qc * Oc);
    hipMemsetAsync(d_out, 0, (size_t)Bc * Oc * sizeof(float), stream);

    psi_kernel<<<dim3(Qc, Bc / 64), 128, 0, stream>>>(
        x, psi_w1b1s, psi_w2s, psi_b2, psi_w3, psi_b3, s_buf);
    phi_kernel<<<dim3(Bc / 64, 8), 128, 0, stream>>>(
        s_buf, phi_w1b1s, phi_w2s, phi_b2, phi_w3, phi_b3, (float*)d_out);
}
